// Quantized4bitLinear_70866960384115
// MI455X (gfx1250) — matrix operand order repaired and checked by reading, not compiled, not run on hardware
//
#include <hip/hip_runtime.h>
#include <stdint.h>

typedef __attribute__((ext_vector_type(16))) __bf16 v16bf;
typedef __attribute__((ext_vector_type(8)))  __bf16 v8bf;
typedef __attribute__((ext_vector_type(8)))  float  v8f;
typedef __attribute__((ext_vector_type(4)))  unsigned int v4u;
typedef __attribute__((ext_vector_type(8)))  int v8i;
typedef __attribute__((ext_vector_type(4)))  int v4i;

__constant__ float NF4_TAB[16] = {
    -1.0f, -0.6961928009986877f, -0.5250730514526367f, -0.39491748809814453f,
    -0.28444138169288635f, -0.18477343022823334f, -0.09105003625154495f, 0.0f,
    0.07958029955625534f, 0.16093020141124725f, 0.24611230194568634f,
    0.33791524171829224f, 0.44070982933044434f, 0.5626170039176941f,
    0.6848514676094055f, 1.0f};

// ---------------------------------------------------------------------------
// Kernel 1: NF4 codes (int32, one 0..15 code per element) -> bf16 weights.
// ---------------------------------------------------------------------------
__global__ __launch_bounds__(256) void nf4_dequant_w(
    const int* __restrict__ codes, const float* __restrict__ absmax,
    __bf16* __restrict__ wq, long long n) {
  long long base = ((long long)blockIdx.x * blockDim.x + threadIdx.x) * 8;
  if (base + 8 > n) return;
  const int4* cp = (const int4*)(codes + base);
  int4 c0 = cp[0];
  int4 c1 = cp[1];
  float s = absmax[base >> 6];
  v8bf o;
  o[0] = (__bf16)(NF4_TAB[c0.x & 15] * s);
  o[1] = (__bf16)(NF4_TAB[c0.y & 15] * s);
  o[2] = (__bf16)(NF4_TAB[c0.z & 15] * s);
  o[3] = (__bf16)(NF4_TAB[c0.w & 15] * s);
  o[4] = (__bf16)(NF4_TAB[c1.x & 15] * s);
  o[5] = (__bf16)(NF4_TAB[c1.y & 15] * s);
  o[6] = (__bf16)(NF4_TAB[c1.z & 15] * s);
  o[7] = (__bf16)(NF4_TAB[c1.w & 15] * s);
  *(v8bf*)(wq + base) = o;
}

// ---------------------------------------------------------------------------
// Kernel 2: out[M,N] = X[M,K] (fp32) * Wq[N,K]^T (bf16) + bias(NF4)
// BM=BN=128, BK=32, double-buffered LDS, 8 waves (2x4), 4x2 WMMA accs/wave.
// B tile: ONE Tensor-Data-Mover descriptor per iteration (wave 0 issues,
// TENSORcnt-fenced), with D# padding fields generating the LDS row padding.
// A tile: VGPR staging for the fp32->bf16 convert.
// ---------------------------------------------------------------------------
#define BM 128
#define BN 128
#define BK 32
#define PAD 8  // 16B row padding (also encoded in the TDM descriptor)

__global__ __launch_bounds__(256, 1) void nf4_gemm(
    const float* __restrict__ X, const __bf16* __restrict__ Wq,
    const int* __restrict__ bcodes, const float* __restrict__ babsmax,
    float* __restrict__ out, int M, int N, int K) {
  __shared__ __bf16 sA[2][BM][BK + PAD];
  __shared__ __bf16 sB[2][BN][BK + PAD];

  const int tid    = threadIdx.x;
  const int lane   = tid & 31;
  const int wid    = tid >> 5;
  const int wave_m = wid >> 2;   // 0..1  -> 64 rows each
  const int wave_n = wid & 3;    // 0..3  -> 32 cols each
  const int half   = lane >> 4;  // ISA half-wave selector
  const int l16    = lane & 15;

  const int tile_m = blockIdx.y * BM;
  const int tile_n = blockIdx.x * BN;

  // A staging: each thread owns row (tid>>1), 16-elem chunk (tid&1).
  const int ldr = tid >> 1;
  const int ldc = (tid & 1) * 16;
  const float* Arow = X + (long long)(tile_m + ldr) * K + ldc;

  auto stageA = [&](int buf, int k0) {
    float4 f0 = *(const float4*)(Arow + k0 + 0);
    float4 f1 = *(const float4*)(Arow + k0 + 4);
    float4 f2 = *(const float4*)(Arow + k0 + 8);
    float4 f3 = *(const float4*)(Arow + k0 + 12);
    __bf16* da = &sA[buf][ldr][ldc];
    da[0]  = (__bf16)f0.x; da[1]  = (__bf16)f0.y;
    da[2]  = (__bf16)f0.z; da[3]  = (__bf16)f0.w;
    da[4]  = (__bf16)f1.x; da[5]  = (__bf16)f1.y;
    da[6]  = (__bf16)f1.z; da[7]  = (__bf16)f1.w;
    da[8]  = (__bf16)f2.x; da[9]  = (__bf16)f2.y;
    da[10] = (__bf16)f2.z; da[11] = (__bf16)f2.w;
    da[12] = (__bf16)f3.x; da[13] = (__bf16)f3.y;
    da[14] = (__bf16)f3.z; da[15] = (__bf16)f3.w;
  };

  // B staging: one TDM descriptor moves the whole 128x32 bf16 tile to LDS.
  // D# group0: count=1 | lds_addr | global_addr[56:0] | type=2.
  // D# group1: data_size=2B, pad every 64B by 16B, dims/strides for [N,K].
  auto stageB = [&](int buf, int k0) {
    if (wid != 0) return;
    unsigned long long ga =
        (unsigned long long)(const void*)(Wq + (long long)tile_n * K + k0);
    unsigned lds = (unsigned)(unsigned long long)(const void*)&sB[buf][0][0];
    v4u g0;
    g0[0] = 1u;                                   // count=1 (valid user D#)
    g0[1] = lds;                                  // lds_addr
    g0[2] = (unsigned)(ga & 0xFFFFFFFFu);         // global_addr[31:0]
    g0[3] = (unsigned)((ga >> 32) & 0x1FFFFFFu)   // global_addr[56:32]
            | (2u << 30);                         // type=2 ("image")
    v8i g1;
    g1[0] = (int)((1u << 16)      // data_size = 2 bytes
                | (1u << 20)      // pad_enable
                | (3u << 22)      // pad_interval: every 16 DWORDs (64B)
                | (3u << 25));    // pad_amount: 4 DWORDs (16B)
    g1[1] = (int)(((unsigned)K & 0xFFFFu) << 16);                 // dim0 lo16
    g1[2] = (int)((((unsigned)K >> 16) & 0xFFFFu)
                | (((unsigned)N & 0xFFFFu) << 16));               // dim0 hi|dim1 lo
    g1[3] = (int)((((unsigned)N >> 16) & 0xFFFFu)
                | ((unsigned)BK << 16));                          // dim1 hi|tile_dim0
    g1[4] = (int)(unsigned)BN;            // tile_dim1 (tile_dim2 = 0)
    g1[5] = (int)(unsigned)K;             // tensor_dim0_stride lo32
    g1[6] = 0;                            // stride hi16 | dim1_stride lo16
    g1[7] = 0;                            // dim1_stride (unused, 2D tile)
    v4i z4 = {0, 0, 0, 0};
#if __has_builtin(__builtin_amdgcn_tensor_load_to_lds)
#if __clang_major__ >= 23
    v8i z8 = {0, 0, 0, 0, 0, 0, 0, 0};
    __builtin_amdgcn_tensor_load_to_lds(g0, g1, z4, z4, z8, 0);
#else
    __builtin_amdgcn_tensor_load_to_lds(g0, g1, z4, z4, 0);
#endif
#else
    asm volatile("tensor_load_to_lds %0, %1" :: "s"(g0), "s"(g1) : "memory");
    (void)z4;
#endif
  };

  v8f acc[4][2];
#pragma unroll
  for (int i = 0; i < 4; ++i)
#pragma unroll
    for (int j = 0; j < 2; ++j)
#pragma unroll
      for (int r = 0; r < 8; ++r) acc[i][j][r] = 0.0f;

  // Prologue: stage tile 0 into buffer 0.
  stageA(0, 0);
  stageB(0, 0);
  if (wid == 0) __builtin_amdgcn_s_wait_tensorcnt(0);
  __syncthreads();

  for (int k0 = 0; k0 < K; k0 += BK) {
    const int cur = (k0 >> 5) & 1;
    const int nxt = cur ^ 1;

    // Stage next tile into the other buffer while computing on this one.
    if (k0 + BK < K) {
      stageA(nxt, k0 + BK);
      stageB(nxt, k0 + BK);
      if (k0 + 2 * BK < K) __builtin_prefetch(Arow + k0 + 2 * BK, 0, 3);
    }

    // --- B fragments: K = half*8..+7 and half*8+16..+23 per lane (ISA B layout) ---
    v16bf bfrag[2];
#pragma unroll
    for (int tn = 0; tn < 2; ++tn) {
      const __bf16* p = &sB[cur][wave_n * 32 + tn * 16 + l16][half * 8];
      v8bf lo = *(const v8bf*)p;
      v8bf hi = *(const v8bf*)(p + 16);
#pragma unroll
      for (int i = 0; i < 8; ++i) { bfrag[tn][i] = lo[i]; bfrag[tn][i + 8] = hi[i]; }
    }

    // --- A fragments + WMMA ---
#pragma unroll
    for (int tm = 0; tm < 4; ++tm) {
      const __bf16* pa = &sA[cur][wave_m * 64 + tm * 16 + l16][half * 8];
      v8bf alo = *(const v8bf*)pa;          // K = half*8 .. +7
      v8bf ahi = *(const v8bf*)(pa + 16);   // K = half*8+16 .. +23
      v16bf a;
#pragma unroll
      for (int i = 0; i < 8; ++i) { a[i] = alo[i]; a[i + 8] = ahi[i]; }
#pragma unroll
      for (int tn = 0; tn < 2; ++tn) {
        acc[tm][tn] = __builtin_amdgcn_wmma_f32_16x16x32_bf16(
            false, a, false, bfrag[tn], (short)0, acc[tm][tn], false, false);
      }
    }

    // Publish buffer `nxt`: TDM fence (wave 0) + workgroup barrier.
    if (wid == 0) __builtin_amdgcn_s_wait_tensorcnt(0);
    __syncthreads();
  }

  // --- epilogue: add NF4-dequantized bias, store fp32 ---
#pragma unroll
  for (int tn = 0; tn < 2; ++tn) {
    int n = tile_n + wave_n * 32 + tn * 16 + l16;
    float bias = NF4_TAB[bcodes[n] & 15] * babsmax[n >> 6];
#pragma unroll
    for (int tm = 0; tm < 4; ++tm) {
      int m0 = tile_m + wave_m * 64 + tm * 16 + half * 8;
      float* op = out + (long long)m0 * N + n;
#pragma unroll
      for (int r = 0; r < 8; ++r) op[(long long)r * N] = acc[tm][tn][r] + bias;
    }
  }
}

// ---------------------------------------------------------------------------
extern "C" void kernel_launch(void* const* d_in, const int* in_sizes, int n_in,
                              void* d_out, int out_size, void* d_ws,
                              size_t ws_size, hipStream_t stream) {
  const float* x        = (const float*)d_in[0];
  const int*   w_codes  = (const int*)d_in[1];
  const float* w_absmax = (const float*)d_in[2];
  const int*   b_codes  = (const int*)d_in[3];
  const float* b_absmax = (const float*)d_in[4];
  float*       out      = (float*)d_out;

  const long long NK = in_sizes[1];          // D_OUT * D_IN
  const int N = in_sizes[3];                 // D_OUT
  const int K = (int)(NK / N);               // D_IN
  const int M = (int)((long long)in_sizes[0] / K);  // B * S

  __bf16* wq = (__bf16*)d_ws;                // 2*NK bytes of scratch

  long long dq_threads = NK / 8;
  dim3 dq_grid((unsigned)((dq_threads + 255) / 256));
  nf4_dequant_w<<<dq_grid, 256, 0, stream>>>(w_codes, w_absmax, wq, NK);

  dim3 grid((unsigned)(N / BN), (unsigned)(M / BM));
  nf4_gemm<<<grid, 256, 0, stream>>>(x, wq, b_codes, b_absmax, out, M, N, K);
}
